// LDMNet_36180804501771
// MI455X (gfx1250) — matrix operand order repaired and checked by reading, not compiled, not run on hardware
//
#include <hip/hip_runtime.h>

#define Bn 4096
#define Fd 256
#define B2 8191

typedef __attribute__((ext_vector_type(16))) __bf16 v16bf;
typedef __attribute__((ext_vector_type(8)))  float  v8f;

union Frag { v16bf v; uint4 q[2]; };

__device__ __forceinline__ unsigned short f2bf(float x){
  unsigned u = __float_as_uint(x);
  unsigned r = u + 0x7FFFu + ((u >> 16) & 1u);
  return (unsigned short)(r >> 16);
}
__device__ __forceinline__ float bf2f(unsigned short h){
  return __uint_as_float(((unsigned)h) << 16);
}
// orderable encoding for float atomicMax on uint
__device__ __forceinline__ unsigned fenc(float f){
  unsigned u = __float_as_uint(f);
  return (u & 0x80000000u) ? ~u : (u | 0x80000000u);
}
__device__ __forceinline__ float fdec(unsigned e){
  unsigned u = (e & 0x80000000u) ? (e & 0x7FFFFFFFu) : ~e;
  return __uint_as_float(u);
}

__device__ __forceinline__ v8f wmma_bf16(v16bf a, v16bf b, v8f c){
  return __builtin_amdgcn_wmma_f32_16x16x32_bf16(false, a, false, b, (short)0, c, false, false);
}

// A fragment 16x32 bf16: lane holds row m=lane&15; lanes<16 K {0..7,16..23}, lanes>=16 K {8..15,24..31}
__device__ __forceinline__ void loadA(Frag& f, const unsigned short* base, int row, int k0, int lane){
  const unsigned short* p = base + row*Fd + k0 + ((lane>>4)<<3);
  f.q[0] = *(const uint4*)(p);
  f.q[1] = *(const uint4*)(p + 16);
}
// B fragment 32x16 bf16 (B[k][n] = src[n][k]): lane holds col n=lane&15; lanes<16 K {0..7,16..23}, lanes>=16 K {8..15,24..31}
__device__ __forceinline__ void loadB(Frag& f, const unsigned short* base, int col, int k0, int lane){
  const unsigned short* p = base + col*Fd + k0 + ((lane>>4)<<3);
  f.q[0] = *(const uint4*)(p);
  f.q[1] = *(const uint4*)(p + 16);
}

__device__ __forceinline__ float bred64(float v, float* buf, int t){
  __syncthreads();
  buf[t] = v; __syncthreads();
  if (t < 32) buf[t] += buf[t+32]; __syncthreads();
  if (t < 16) buf[t] += buf[t+16]; __syncthreads();
  if (t <  8) buf[t] += buf[t+ 8]; __syncthreads();
  if (t <  4) buf[t] += buf[t+ 4]; __syncthreads();
  if (t <  2) buf[t] += buf[t+ 2]; __syncthreads();
  if (t <  1) buf[0] += buf[1];    __syncthreads();
  return buf[0];
}

// ---------------- Kernel 1: L2-normalize rows, bf16 hi/lo split, per-row scalars ----------------
__global__ __launch_bounds__(64) void k_norm(
    const float* __restrict__ p, const float* __restrict__ a,
    float* __restrict__ pe, float* __restrict__ ae,
    unsigned short* __restrict__ peh, unsigned short* __restrict__ pel,
    unsigned short* __restrict__ aeh, unsigned short* __restrict__ ael,
    float* __restrict__ cosm, float* __restrict__ dpa,
    unsigned* __restrict__ rmaxE, unsigned* __restrict__ cmaxE)
{
  __shared__ float buf[64];
  int row = blockIdx.x, t = threadIdx.x;
  float4 pv = ((const float4*)(p + row*Fd))[t];
  float4 av = ((const float4*)(a + row*Fd))[t];
  float sp = pv.x*pv.x + pv.y*pv.y + pv.z*pv.z + pv.w*pv.w;
  float sa = av.x*av.x + av.y*av.y + av.z*av.z + av.w*av.w;
  sp = bred64(sp, buf, t);
  sa = bred64(sa, buf, t);
  float ip = 1.0f / sqrtf(sp);
  float ia = 1.0f / sqrtf(sa);
  float pn[4] = { pv.x*ip, pv.y*ip, pv.z*ip, pv.w*ip };
  float an[4] = { av.x*ia, av.y*ia, av.z*ia, av.w*ia };

  int base = row*Fd + t*4;
  float dt = 0.f, dd = 0.f;
  ushort4 ph4, pl4, ah4, al4;
  unsigned short h, l;
  #pragma unroll
  for (int c = 0; c < 4; c++){
    pe[base + c] = pn[c];
    ae[base + c] = an[c];
    float dx = pn[c] - an[c];
    dt += pn[c]*an[c];
    dd += dx*dx;
  }
  h = f2bf(pn[0]); ph4.x = h; pl4.x = f2bf(pn[0]-bf2f(h));
  h = f2bf(pn[1]); ph4.y = h; pl4.y = f2bf(pn[1]-bf2f(h));
  h = f2bf(pn[2]); ph4.z = h; pl4.z = f2bf(pn[2]-bf2f(h));
  h = f2bf(pn[3]); ph4.w = h; pl4.w = f2bf(pn[3]-bf2f(h));
  h = f2bf(an[0]); ah4.x = h; al4.x = f2bf(an[0]-bf2f(h));
  h = f2bf(an[1]); ah4.y = h; al4.y = f2bf(an[1]-bf2f(h));
  h = f2bf(an[2]); ah4.z = h; al4.z = f2bf(an[2]-bf2f(h));
  h = f2bf(an[3]); ah4.w = h; al4.w = f2bf(an[3]-bf2f(h));
  (void)l;
  ((ushort4*)(peh + row*Fd))[t] = ph4;
  ((ushort4*)(pel + row*Fd))[t] = pl4;
  ((ushort4*)(aeh + row*Fd))[t] = ah4;
  ((ushort4*)(ael + row*Fd))[t] = al4;

  dt = bred64(dt, buf, t);
  dd = bred64(dd, buf, t);
  if (t == 0){
    const float COSM = 0.87758256189037271612f;
    const float SINM = 0.47942553860420300027f;
    const float MMC  = 0.23971276930210150013f;
    const float THRESH = -0.87758256189037271612f;
    float c = fminf(1.f, fmaxf(-1.f, dt));
    float s = sqrtf(fmaxf(0.f, 1.f - c*c));
    cosm[row] = (c - THRESH <= 0.f) ? (c - MMC) : (c*COSM - s*SINM);
    dpa[row]  = sqrtf(dd);
    rmaxE[row] = fenc(-2.f);
    cmaxE[row] = fenc(-2.f);
  }
}

// ---------------- Kernel 2: g = pe @ ae^T, write angles (both halves), row/col maxima ----------------
__global__ __launch_bounds__(128) void k_gemm_angles(
    const unsigned short* __restrict__ peh, const unsigned short* __restrict__ pel,
    const unsigned short* __restrict__ aeh, const unsigned short* __restrict__ ael,
    const float* __restrict__ cosm,
    unsigned* __restrict__ rmaxE, unsigned* __restrict__ cmaxE,
    float* __restrict__ angles)
{
  __shared__ float lds[64*65];
  int lane = threadIdx.x & 31;
  int wave = threadIdx.x >> 5;
  int wr = (wave >> 1) * 32, wc = (wave & 1) * 32;
  int I0 = blockIdx.y * 64, J0 = blockIdx.x * 64;
  int n_l = lane & 15;
  int mo = (lane >> 4) * 8;

  v8f C[2][2] = {};
  for (int k0 = 0; k0 < Fd; k0 += 32){
    Frag ah[2], al[2], bh[2], bl[2];
    #pragma unroll
    for (int t = 0; t < 2; t++){
      int r = I0 + wr + t*16 + n_l;
      int c = J0 + wc + t*16 + n_l;
      loadA(ah[t], peh, r, k0, lane);
      loadA(al[t], pel, r, k0, lane);
      loadB(bh[t], aeh, c, k0, lane);
      loadB(bl[t], ael, c, k0, lane);
    }
    #pragma unroll
    for (int ti = 0; ti < 2; ti++)
      #pragma unroll
      for (int tj = 0; tj < 2; tj++){
        C[ti][tj] = wmma_bf16(ah[ti].v, bh[tj].v, C[ti][tj]);
        C[ti][tj] = wmma_bf16(ah[ti].v, bl[tj].v, C[ti][tj]);
        C[ti][tj] = wmma_bf16(al[ti].v, bh[tj].v, C[ti][tj]);
      }
  }

  float rowm[2][8];
  #pragma unroll
  for (int ti = 0; ti < 2; ti++)
    #pragma unroll
    for (int v = 0; v < 8; v++) rowm[ti][v] = -2.f;
  float colm[2] = { -2.f, -2.f };

  #pragma unroll
  for (int ti = 0; ti < 2; ti++){
    #pragma unroll
    for (int tj = 0; tj < 2; tj++){
      #pragma unroll
      for (int v = 0; v < 8; v++){
        float g = C[ti][tj][v];
        int il = wr + ti*16 + mo + v;
        int jl = wc + tj*16 + n_l;
        int i = I0 + il, j = J0 + jl;
        float cl = fminf(1.f, fmaxf(-1.f, g));
        float a1 = (i == j) ? 64.f * cosm[i] : 64.f * cl;
        angles[(size_t)i * B2 + j] = a1;
        lds[jl*65 + il] = cl;
        float gm = (i == j) ? -2.f : g;
        rowm[ti][v] = fmaxf(rowm[ti][v], gm);
        colm[tj]   = fmaxf(colm[tj], gm);
      }
    }
  }
  // row maxima: reduce across 16 lanes holding the row
  #pragma unroll
  for (int ti = 0; ti < 2; ti++){
    #pragma unroll
    for (int v = 0; v < 8; v++){
      float r = rowm[ti][v];
      for (int m = 1; m < 16; m <<= 1) r = fmaxf(r, __shfl_xor(r, m));
      if (n_l == 0){
        int i = I0 + wr + ti*16 + mo + v;
        atomicMax(&rmaxE[i], fenc(r));
      }
    }
  }
  // col maxima
  #pragma unroll
  for (int tj = 0; tj < 2; tj++){
    float c = colm[tj];
    c = fmaxf(c, __shfl_xor(c, 16));
    if (lane < 16){
      int j = J0 + wc + tj*16 + n_l;
      atomicMax(&cmaxE[j], fenc(c));
    }
  }
  __syncthreads();
  // transposed, diagonal-removed write: angles[j, 4096 + (i<j?i:i-1)]
  int jl = threadIdx.x >> 1;
  int j = J0 + jl;
  int ibase = (threadIdx.x & 1) * 32;
  for (int s = 0; s < 32; s++){
    int il = ibase + s;
    int i = I0 + il;
    if (i == j) continue;
    int k = (i < j) ? i : i - 1;
    angles[(size_t)j * B2 + Bn + k] = 64.f * lds[jl*65 + il];
  }
}

// ---------------- Kernel 3: fused ae@ae^T and pe@pe^T, per-row candidate mining ----------------
__global__ __launch_bounds__(128) void k_gemm_mine(
    const unsigned short* __restrict__ peh, const unsigned short* __restrict__ pel,
    const unsigned short* __restrict__ aeh, const unsigned short* __restrict__ ael,
    float* __restrict__ cA_g, float* __restrict__ cA_p, int* __restrict__ cA_j,
    float* __restrict__ cP_a, float* __restrict__ cP_p, int* __restrict__ cP_j)
{
  __shared__ float S_ag[64][2], S_ap[64][2], S_pa[64][2], S_pp[64][2];
  __shared__ int   S_aj[64][2], S_pj[64][2];

  int lane = threadIdx.x & 31;
  int wave = threadIdx.x >> 5;
  int wr = (wave >> 1) * 32, wc = (wave & 1) * 32;
  int half = wave & 1;
  int I0 = blockIdx.y * 64, J0 = blockIdx.x * 64;
  int n_l = lane & 15;
  int mo = (lane >> 4) * 8;

  v8f CA[2][2] = {};
  v8f CP[2][2] = {};
  for (int k0 = 0; k0 < Fd; k0 += 32){
    Frag Aah[2], Aal[2], Aph[2], Apl[2], Bah[2], Bal[2], Bph[2], Bpl[2];
    #pragma unroll
    for (int t = 0; t < 2; t++){
      int r = I0 + wr + t*16 + n_l;
      int c = J0 + wc + t*16 + n_l;
      loadA(Aah[t], aeh, r, k0, lane);
      loadA(Aal[t], ael, r, k0, lane);
      loadA(Aph[t], peh, r, k0, lane);
      loadA(Apl[t], pel, r, k0, lane);
      loadB(Bah[t], aeh, c, k0, lane);
      loadB(Bal[t], ael, c, k0, lane);
      loadB(Bph[t], peh, c, k0, lane);
      loadB(Bpl[t], pel, c, k0, lane);
    }
    #pragma unroll
    for (int ti = 0; ti < 2; ti++)
      #pragma unroll
      for (int tj = 0; tj < 2; tj++){
        CA[ti][tj] = wmma_bf16(Aah[ti].v, Bah[tj].v, CA[ti][tj]);
        CA[ti][tj] = wmma_bf16(Aah[ti].v, Bal[tj].v, CA[ti][tj]);
        CA[ti][tj] = wmma_bf16(Aal[ti].v, Bah[tj].v, CA[ti][tj]);
        CP[ti][tj] = wmma_bf16(Aph[ti].v, Bph[tj].v, CP[ti][tj]);
        CP[ti][tj] = wmma_bf16(Aph[ti].v, Bpl[tj].v, CP[ti][tj]);
        CP[ti][tj] = wmma_bf16(Apl[ti].v, Bph[tj].v, CP[ti][tj]);
      }
  }

  #pragma unroll
  for (int ti = 0; ti < 2; ti++){
    #pragma unroll
    for (int v = 0; v < 8; v++){
      int i = I0 + wr + ti*16 + mo + v;
      float aAg, aAp, pAg, pAp;
      int aAj, pAj;
      {
        float gA = CA[ti][0][v], gP = CP[ti][0][v];
        int j = J0 + wc + n_l;
        if (i == j){ gA = -2.f; gP = -2.f; }
        aAg = gA; aAp = gP; aAj = j;
        pAg = gA; pAp = gP; pAj = j;
      }
      {
        float gA = CA[ti][1][v], gP = CP[ti][1][v];
        int j = J0 + wc + 16 + n_l;
        if (i == j){ gA = -2.f; gP = -2.f; }
        if (gA > aAg){ aAg = gA; aAp = gP; aAj = j; }
        if (gP > pAp){ pAg = gA; pAp = gP; pAj = j; }
      }
      for (int m = 1; m < 16; m <<= 1){
        float oAg = __shfl_xor(aAg, m), oAp = __shfl_xor(aAp, m);
        int   oAj = __shfl_xor(aAj, m);
        if (oAg > aAg){ aAg = oAg; aAp = oAp; aAj = oAj; }
        float oPg = __shfl_xor(pAg, m), oPp = __shfl_xor(pAp, m);
        int   oPj = __shfl_xor(pAj, m);
        if (oPp > pAp){ pAg = oPg; pAp = oPp; pAj = oPj; }
      }
      if (n_l == 0){
        int r = wr + ti*16 + mo + v;
        S_ag[r][half] = aAg; S_ap[r][half] = aAp; S_aj[r][half] = aAj;
        S_pa[r][half] = pAg; S_pp[r][half] = pAp; S_pj[r][half] = pAj;
      }
    }
  }
  __syncthreads();
  if (threadIdx.x < 64){
    int r = threadIdx.x;
    int i = I0 + r;
    int tJ = blockIdx.x;
    size_t o = (size_t)i * 64 + tJ;
    int w  = (S_ag[r][1] > S_ag[r][0]) ? 1 : 0;
    cA_g[o] = S_ag[r][w]; cA_p[o] = S_ap[r][w]; cA_j[o] = S_aj[r][w];
    int w2 = (S_pp[r][1] > S_pp[r][0]) ? 1 : 0;
    cP_a[o] = S_pa[r][w2]; cP_p[o] = S_pp[r][w2]; cP_j[o] = S_pj[r][w2];
  }
}

// ---------------- Kernel 4a: per-row SOS + hard-neg terms ----------------
__global__ __launch_bounds__(128) void k_rowstats(
    const float* __restrict__ cA_g, const float* __restrict__ cA_p, const int* __restrict__ cA_j,
    const float* __restrict__ cP_a, const float* __restrict__ cP_p, const int* __restrict__ cP_j,
    const unsigned* __restrict__ rmaxE, const unsigned* __restrict__ cmaxE,
    const float* __restrict__ dpa,
    float* __restrict__ rowS, float* __restrict__ rowD)
{
  int i = blockIdx.x * blockDim.x + threadIdx.x;
  if (i >= Bn) return;
  float bAg = -3.f, bAp = 0.f; int bAj = -1;
  float bPp = -3.f, bPa = 0.f; int bPj = -2;
  for (int t = 0; t < 64; t++){
    size_t o = (size_t)i * 64 + t;
    float g = cA_g[o];
    if (g > bAg){ bAg = g; bAp = cA_p[o]; bAj = cA_j[o]; }
    float pp = cP_p[o];
    if (pp > bPp){ bPp = pp; bPa = cP_a[o]; bPj = cP_j[o]; }
  }
  float DA1 = sqrtf(fmaxf(2.f - 2.f*bAg, 1e-12f));
  float DP1 = sqrtf(fmaxf(2.f - 2.f*bAp, 1e-12f));
  float s = (DA1 - DP1) * (DA1 - DP1);
  if (bAj != bPj){
    float DA2 = sqrtf(fmaxf(2.f - 2.f*bPa, 1e-12f));
    float DP2 = sqrtf(fmaxf(2.f - 2.f*bPp, 1e-12f));
    s += (DA2 - DP2) * (DA2 - DP2);
  }
  rowS[i] = sqrtf(s);
  float g = fmaxf(fdec(rmaxE[i]), fdec(cmaxE[i]));
  float dn = sqrtf(fmaxf(2.f - 2.f*g, 1e-12f));
  float df = fmaxf(dpa[i] - dn + 1.0f, 0.f);
  rowD[i] = df * df;
}

// ---------------- Kernel 4b: deterministic final scalar reduction ----------------
__global__ __launch_bounds__(1024) void k_final(
    const float* __restrict__ rowS, const float* __restrict__ rowD, float* __restrict__ out)
{
  __shared__ float buf[1024];
  float s = 0.f;
  for (int i = threadIdx.x; i < Bn; i += 1024) s += rowS[i] + rowD[i];
  buf[threadIdx.x] = s; __syncthreads();
  for (int st = 512; st > 0; st >>= 1){
    if (threadIdx.x < st) buf[threadIdx.x] += buf[threadIdx.x + st];
    __syncthreads();
  }
  if (threadIdx.x == 0) out[0] = buf[0] * (1.0f / (float)Bn);
}

extern "C" void kernel_launch(void* const* d_in, const int* in_sizes, int n_in,
                              void* d_out, int out_size, void* d_ws, size_t ws_size,
                              hipStream_t stream) {
  (void)in_sizes; (void)n_in; (void)out_size; (void)ws_size;
  const float* p = (const float*)d_in[0];
  const float* a = (const float*)d_in[1];
  // d_in[2] (label) unused: BETA == 0 and angles don't depend on it.

  float* out    = (float*)d_out;
  float* angles = out + 1;
  float* pe     = out + 1 + (size_t)Bn * B2;
  float* ae     = pe + (size_t)Bn * Fd;

  char* w = (char*)d_ws;
  unsigned short* peh = (unsigned short*)w; w += (size_t)Bn*Fd*2;
  unsigned short* pel = (unsigned short*)w; w += (size_t)Bn*Fd*2;
  unsigned short* aeh = (unsigned short*)w; w += (size_t)Bn*Fd*2;
  unsigned short* ael = (unsigned short*)w; w += (size_t)Bn*Fd*2;
  float*    cosm  = (float*)w;    w += Bn*4;
  float*    dpa   = (float*)w;    w += Bn*4;
  unsigned* rmaxE = (unsigned*)w; w += Bn*4;
  unsigned* cmaxE = (unsigned*)w; w += Bn*4;
  float*    rowS  = (float*)w;    w += Bn*4;
  float*    rowD  = (float*)w;    w += Bn*4;
  float* cA_g = (float*)w; w += (size_t)Bn*64*4;
  float* cA_p = (float*)w; w += (size_t)Bn*64*4;
  int*   cA_j = (int*)w;   w += (size_t)Bn*64*4;
  float* cP_a = (float*)w; w += (size_t)Bn*64*4;
  float* cP_p = (float*)w; w += (size_t)Bn*64*4;
  int*   cP_j = (int*)w;   w += (size_t)Bn*64*4;

  k_norm<<<Bn, 64, 0, stream>>>(p, a, pe, ae, peh, pel, aeh, ael, cosm, dpa, rmaxE, cmaxE);
  k_gemm_angles<<<dim3(64,64), 128, 0, stream>>>(peh, pel, aeh, ael, cosm, rmaxE, cmaxE, angles);
  k_gemm_mine<<<dim3(64,64), 128, 0, stream>>>(peh, pel, aeh, ael, cA_g, cA_p, cA_j, cP_a, cP_p, cP_j);
  k_rowstats<<<32, 128, 0, stream>>>(cA_g, cA_p, cA_j, cP_a, cP_p, cP_j, rmaxE, cmaxE, dpa, rowS, rowD);
  k_final<<<1, 1024, 0, stream>>>(rowS, rowD, out);
}
